// AggregatedAttention_12197707121248
// MI455X (gfx1250) — matrix operand order repaired and checked by reading, not compiled, not run on hardware
//
#include <hip/hip_runtime.h>
#include <hip/hip_bf16.h>
#include <math.h>

// ---------------------------------------------------------------------------
// Problem constants (match reference setup_inputs: RES=64, DIM=256, B=4)
// ---------------------------------------------------------------------------
constexpr int   Bc     = 4;
constexpr int   Hc     = 64, Wc = 64;
constexpr int   Nn     = Hc * Wc;          // 4096
constexpr int   Cc     = 256;
constexpr int   HEADSc = 8;
constexpr int   HDc    = 32;               // == wave32 lanes
constexpr int   PLc    = 64;               // (H/8)*(W/8)
constexpr int   LOCALc = 9;
constexpr int   TBLc   = 4096;
constexpr int   BN     = Bc * Nn;          // 16384
constexpr float EPSc   = 1.1920929e-07f;

typedef __attribute__((ext_vector_type(16))) __bf16 bf16x16;
typedef __attribute__((ext_vector_type(8)))  __bf16 bf16x8;
typedef __attribute__((ext_vector_type(8)))  float  f32x8;

// ---------------------------------------------------------------------------
// Wave32 helpers
// ---------------------------------------------------------------------------
static __device__ __forceinline__ float wred_add(float v) {
#pragma unroll
  for (int off = 16; off > 0; off >>= 1) v += __shfl_xor(v, off, 32);
  return v;
}
static __device__ __forceinline__ float wred_max(float v) {
#pragma unroll
  for (int off = 16; off > 0; off >>= 1) v = fmaxf(v, __shfl_xor(v, off, 32));
  return v;
}

static __device__ __forceinline__ void wait_async0() {
#if __has_builtin(__builtin_amdgcn_s_wait_asynccnt)
  __builtin_amdgcn_s_wait_asynccnt(0);
#else
  asm volatile("s_wait_asynccnt 0" ::: "memory");
#endif
}

// ---------------------------------------------------------------------------
// f32 -> bf16 conversion
// ---------------------------------------------------------------------------
__global__ __launch_bounds__(256) void cvt_bf16_kernel(const float* __restrict__ in,
                                                       __bf16* __restrict__ out, int n) {
  int i = blockIdx.x * 256 + threadIdx.x;
  if (i < n) out[i] = (__bf16)in[i];
}

// ---------------------------------------------------------------------------
// WMMA bf16 GEMM:  C[M,N] = A[M,K] * W[N,K]^T + bias[N]   (fp32 accumulate)
// K is a compile-time constant (256 here) so the K-loop fully unrolls:
// no pipeline-rotation register copies, no WMMA->VALU hazard nops, and the
// scheduler hoists each step's loads across the previous wmma group.
//
// block = 256 threads = 8 waves; block tile 128M x 64N; wave tile 16M x 64N.
// The 64-column B panel (64 x K bf16 -> 32KB) is shared by all 8 waves, so it
// is staged ONCE into LDS with CDNA5 async global->LDS DMA
// (global_load_async_to_lds_b128, ASYNCcnt), then fragments come from
// ds_load_b128. ldsB is passed as a dummy asm operand so the shared buffer
// escapes into the asm and the memory clobber covers it.
//
// A fragment (16x32 bf16): lane l holds row M=l&15; K elems {b0..b0+7, b1..b1+7}
// with b0 = k0 + (l<16?0:8), b1 = k0 + (l<16?16:24)  -> two b128 loads.
// B fragment (32x16 bf16): lane l holds col N=l&15; same K elems as A fragment
// ---------------------------------------------------------------------------
constexpr int LDS_STR = 264;   // bf16 elems per LDS row: 132 dwords -> rows 4 banks apart

static __device__ __forceinline__ bf16x16 cat8(bf16x8 lo, bf16x8 hi) {
  return __builtin_shufflevector(lo, hi, 0, 1, 2, 3, 4, 5, 6, 7,
                                 8, 9, 10, 11, 12, 13, 14, 15);
}

static __device__ __forceinline__ bf16x16 load_afrag(const __bf16* ap) {
  bf16x8 lo = *(const bf16x8*)ap;
  bf16x8 hi = *(const bf16x8*)(ap + 16);
  return cat8(lo, hi);
}

template <int K>
__global__ __launch_bounds__(256)
void wmma_gemm_bf16(const __bf16* __restrict__ A, const __bf16* __restrict__ Bw,
                    const float* __restrict__ bias, float* __restrict__ C,
                    int M, int N) {
  __shared__ __bf16 ldsB[64 * LDS_STR];

  const int lane   = threadIdx.x & 31;
  const int wave   = threadIdx.x >> 5;              // 0..7
  const int mBase  = blockIdx.x * 128 + wave * 16;
  const int nBase  = blockIdx.y * 64;
  const bool hiHalf = lane >= 16;
  const int  mRow   = mBase + (lane & 15);

  // ---- async-stage B panel: 64 rows x K bf16, 16B chunks -----------------
  constexpr int ChPerRow = K >> 3;                  // 16B chunks per row
  constexpr int nChunks  = 64 * ChPerRow;
#pragma unroll
  for (int c0 = 0; c0 < nChunks; c0 += 256) {
    const int c   = c0 + threadIdx.x;
    const int row = c / ChPerRow;
    const int kof = (c % ChPerRow) * 8;
    const __bf16* src = Bw + (size_t)(nBase + row) * K + kof;
    const unsigned dst = (unsigned)((row * LDS_STR + kof) * (int)sizeof(__bf16));
    // %2 (ldsB) is an escape operand: makes the shared buffer reachable from
    // the asm so the "memory" clobber covers the async LDS writes.
    asm volatile("global_load_async_to_lds_b128 %0, %1, off"
                 :: "v"(dst), "v"(src), "s"(ldsB)
                 : "memory");
  }
  wait_async0();
  __syncthreads();

  f32x8 zero = {0.f, 0.f, 0.f, 0.f, 0.f, 0.f, 0.f, 0.f};
  f32x8 acc[4];
#pragma unroll
  for (int t = 0; t < 4; ++t) acc[t] = zero;

  const __bf16* aPtr = A + (size_t)mRow * K + (hiHalf ? 8 : 0);

#pragma unroll
  for (int k0 = 0; k0 < K; k0 += 32) {
    bf16x16 af = load_afrag(aPtr + k0);
    bf16x16 bfr[4];
#pragma unroll
    for (int t = 0; t < 4; ++t) {
      const __bf16* bp = &ldsB[(t * 16 + (lane & 15)) * LDS_STR + k0 + (hiHalf ? 8 : 0)];
      bf16x8 b0 = *(const bf16x8*)bp;
      bf16x8 b1 = *(const bf16x8*)(bp + 16);
      bfr[t] = cat8(b0, b1);
    }
#pragma unroll
    for (int t = 0; t < 4; ++t)
      acc[t] = __builtin_amdgcn_wmma_f32_16x16x32_bf16(false, af, false, bfr[t],
                                                       (short)0, acc[t], false, false);
  }

  // C/D layout: lane l -> N = nTile + (l&15); VGPR r -> M = mBase + r + (l<16?0:8)
#pragma unroll
  for (int t = 0; t < 4; ++t) {
    const int nCol = nBase + t * 16 + (lane & 15);
    const float bv = bias[nCol];
#pragma unroll
    for (int r = 0; r < 8; ++r) {
      const int m = mBase + r + (hiHalf ? 8 : 0);
      C[(size_t)m * N + nCol] = acc[t][r] + bv;
    }
  }
}

// ---------------------------------------------------------------------------
// MoE gating: per token (wave) compute 10 dot products against gate weights,
// softmaxes, top-2 routing -> masked[b,n,8] per-head weights.
// ---------------------------------------------------------------------------
static __device__ __forceinline__ void softmax4(const float* d, float* g) {
  float m = fmaxf(fmaxf(d[0], d[1]), fmaxf(d[2], d[3]));
  float s = 0.f;
#pragma unroll
  for (int t = 0; t < 4; ++t) { g[t] = expf(d[t] - m); s += g[t]; }
  float inv = 1.0f / s;
#pragma unroll
  for (int t = 0; t < 4; ++t) g[t] *= inv;
}

__global__ __launch_bounds__(256)
void gates_kernel(const float* __restrict__ x, const float* __restrict__ wg,
                  const float* __restrict__ wg0, const float* __restrict__ wg1,
                  float* __restrict__ maskedw) {
  int gw   = (blockIdx.x * 256 + threadIdx.x) >> 5;
  int lane = threadIdx.x & 31;
  if (gw >= BN) return;
  const float* xr = x + (size_t)gw * Cc + lane * 8;
  float xv[8];
#pragma unroll
  for (int i = 0; i < 8; ++i) xv[i] = xr[i];

  float dots[10];
#pragma unroll
  for (int r = 0; r < 10; ++r) {
    const float* wr = (r < 4) ? wg + r * Cc : (r < 6) ? wg0 + (r - 4) * Cc : wg1 + (r - 6) * Cc;
    const float* wp = wr + lane * 8;
    float s = 0.f;
#pragma unroll
    for (int i = 0; i < 8; ++i) s += xv[i] * wp[i];
    dots[r] = wred_add(s);
  }

  float g[4];  softmax4(dots, g);
  float w0[2];
  {
    float m = fmaxf(dots[4], dots[5]);
    float e0 = expf(dots[4] - m), e1 = expf(dots[5] - m);
    float inv = 2.0f / (e0 + e1);
    w0[0] = e0 * inv; w0[1] = e1 * inv;
  }
  float sh[4]; softmax4(dots + 6, sh);
#pragma unroll
  for (int t = 0; t < 4; ++t) sh[t] *= 4.0f;

  int i1 = 0;
#pragma unroll
  for (int t = 1; t < 4; ++t) if (g[t] > g[i1]) i1 = t;
  int i2 = (i1 == 0) ? 1 : 0;
#pragma unroll
  for (int t = 0; t < 4; ++t) if (t != i1 && g[t] > g[i2]) i2 = t;
  float rs = fmaxf(g[i1] + g[i2], EPSc);
  float rg[4];
#pragma unroll
  for (int t = 0; t < 4; ++t) rg[t] = ((t == i1) || (t == i2)) ? g[t] / rs * 2.0f : 0.0f;

  if (lane < 8) {
    float mv = (lane < 4) ? w0[0] * sh[lane] : w0[1] * rg[lane - 4];
    maskedw[(size_t)gw * 8 + lane] = mv;
  }
}

// ---------------------------------------------------------------------------
// q post: L2-normalize per (b,n,h) over HD (lane=d), write q_norm in place,
// q_scaled = (q_norm + query_embedding) * softplus(temp) * log(win_area + PL)
// ---------------------------------------------------------------------------
__global__ __launch_bounds__(256)
void q_post_kernel(float* __restrict__ qbuf, float* __restrict__ qs,
                   const float* __restrict__ qe, const float* __restrict__ temp) {
  int gw   = (blockIdx.x * 256 + threadIdx.x) >> 5;
  int lane = threadIdx.x & 31;
  if (gw >= BN * HEADSc) return;
  int h  = gw & 7;
  int bn = gw >> 3;
  int n  = bn & (Nn - 1);
  size_t off = (size_t)bn * Cc + h * HDc + lane;
  float v  = qbuf[off];
  float ss = wred_add(v * v);
  float qn = v / fmaxf(sqrtf(ss), EPSc);
  qbuf[off] = qn;
  int i = n >> 6, j = n & 63;
  int ihi = (i + 1 > Hc - 1) ? Hc - 1 : i + 1, ilo = (i - 1 < 0) ? 0 : i - 1;
  int jhi = (j + 1 > Wc - 1) ? Wc - 1 : j + 1, jlo = (j - 1 < 0) ? 0 : j - 1;
  int ch = ihi - ilo + 1, cw = jhi - jlo + 1;
  float sls = logf((float)(ch * cw) + (float)PLc);
  float sp  = log1pf(expf(temp[h]));
  qs[off] = (qn + qe[h * HDc + lane]) * sp * sls;
}

// ---------------------------------------------------------------------------
// L2-normalize k-part of a [rows][2C] kv buffer, per (row,h), lane=d, in place
// ---------------------------------------------------------------------------
__global__ __launch_bounds__(256)
void kvnorm_kernel(float* __restrict__ kv, int rows) {
  int gw   = (blockIdx.x * 256 + threadIdx.x) >> 5;
  int lane = threadIdx.x & 31;
  if (gw >= rows * HEADSc) return;
  int h  = gw & 7;
  int bn = gw >> 3;
  size_t off = (size_t)bn * (2 * Cc) + h * HDc + lane;
  float v  = kv[off];
  float ss = wred_add(v * v);
  kv[off]  = v / fmaxf(sqrtf(ss), EPSc);
}

// ---------------------------------------------------------------------------
// gelu(exact) + 8x8 average pool: sr[b,n,c] -> xpool[b,p,c]
// ---------------------------------------------------------------------------
__global__ __launch_bounds__(256)
void srpool_kernel(const float* __restrict__ sr, float* __restrict__ xp) {
  int idx = blockIdx.x * 256 + threadIdx.x;
  if (idx >= Bc * PLc * Cc) return;
  int c  = idx & 255;
  int bp = idx >> 8;
  int p  = bp & 63;
  int b  = bp >> 6;
  int pi = p >> 3, pj = p & 7;
  float s = 0.f;
  for (int ii = 0; ii < 8; ++ii)
    for (int jj = 0; jj < 8; ++jj) {
      int n   = (pi * 8 + ii) * Wc + pj * 8 + jj;
      float v = sr[((size_t)(b * Nn + n)) * Cc + c];
      s += 0.5f * v * (1.0f + erff(v * 0.70710678118654752f));
    }
  xp[idx] = s * (1.0f / 64.0f);
}

// ---------------------------------------------------------------------------
// LayerNorm over C=256 per pooled token; emit bf16 for the pooled-kv GEMM
// ---------------------------------------------------------------------------
__global__ __launch_bounds__(256)
void ln_kernel(const float* __restrict__ xp, const float* __restrict__ g,
               const float* __restrict__ b, __bf16* __restrict__ out) {
  int row = blockIdx.x;
  int t   = threadIdx.x;
  float v = xp[(size_t)row * Cc + t];
  __shared__ float red[8];
  float s = wred_add(v);
  if ((t & 31) == 0) red[t >> 5] = s;
  __syncthreads();
  float mu = 0.f;
#pragma unroll
  for (int k = 0; k < 8; ++k) mu += red[k];
  mu *= (1.0f / Cc);
  __syncthreads();
  float d  = v - mu;
  float s2 = wred_add(d * d);
  if ((t & 31) == 0) red[t >> 5] = s2;
  __syncthreads();
  float var = 0.f;
#pragma unroll
  for (int k = 0; k < 8; ++k) var += red[k];
  var *= (1.0f / Cc);
  float y = d * rsqrtf(var + 1e-5f) * g[t] + b[t];
  out[(size_t)row * Cc + t] = (__bf16)y;
}

// ---------------------------------------------------------------------------
// CPB MLP table: cpbt[t][h] = relu(rct[t]@cpb1_w^T+b1) @ cpb2_w[h] + b2[h]
// ---------------------------------------------------------------------------
__global__ __launch_bounds__(256)
void cpb_kernel(const float* __restrict__ rct, const float* __restrict__ w1,
                const float* __restrict__ b1, const float* __restrict__ w2,
                const float* __restrict__ b2, float* __restrict__ cpbt) {
  int idx = blockIdx.x * 256 + threadIdx.x;
  if (idx >= TBLc * HEADSc) return;
  int h = idx & 7;
  int t = idx >> 3;
  float r0 = rct[t * 2], r1 = rct[t * 2 + 1];
  float acc = b2[h];
  for (int j = 0; j < 512; ++j) {
    float hj = r0 * w1[j * 2] + r1 * w1[j * 2 + 1] + b1[j];
    hj = fmaxf(hj, 0.0f);
    acc += hj * w2[h * 512 + j];
  }
  cpbt[t * 8 + h] = acc;
}

// ---------------------------------------------------------------------------
// Fused attention: one wave per (b,h,n) query, lane = d (HD==32).
// local 9 scores + 64 pool scores + joint softmax(73) + learnable-token
// correction + both value contractions + per-head MoE gate -> bf16 out.
// ---------------------------------------------------------------------------
__global__ __launch_bounds__(256)
void attn_kernel(const float* __restrict__ qn_buf, const float* __restrict__ qs_buf,
                 const float* __restrict__ kvbuf, const float* __restrict__ pkv,
                 const float* __restrict__ cpbt, const int* __restrict__ rpi,
                 const float* __restrict__ rpb, const float* __restrict__ ltok,
                 const float* __restrict__ lbias, const float* __restrict__ maskedw,
                 __bf16* __restrict__ outbf) {
  int gw   = (blockIdx.x * 256 + threadIdx.x) >> 5;
  int lane = threadIdx.x & 31;
  if (gw >= Bc * HEADSc * Nn) return;
  int n  = gw & (Nn - 1);
  int bh = gw >> 12;
  int h  = bh & 7;
  int b  = bh >> 3;
  int i  = n >> 6, j = n & 63;

  size_t qoff = ((size_t)(b * Nn + n)) * Cc + h * HDc + lane;
  float qs = qs_buf[qoff];
  float qn = qn_buf[qoff];

  float vloc[LOCALc], aloc[LOCALc], tl[LOCALc];
#pragma unroll
  for (int l = 0; l < LOCALc; ++l) {
    int di = l / 3 - 1, dj = l % 3 - 1;
    int ni = i + di, nj = j + dj;
    bool inb = (ni >= 0) && (ni < Hc) && (nj >= 0) && (nj < Wc);
    float kk = 0.f, vv = 0.f;
    if (inb) {
      size_t o = ((size_t)(b * Nn + ni * Wc + nj)) * (2 * Cc) + h * HDc + lane;
      kk = kvbuf[o];
      vv = kvbuf[o + Cc];
    }
    vloc[l] = vv;
    aloc[l] = wred_add(qs * kk) + rpb[h * LOCALc + l];
    tl[l]   = wred_add(qn * ltok[(h * HDc + lane) * LOCALc + l]);
  }

  // pool scores: lane p%32 keeps score for key p (sc0: p<32, sc1: p>=32)
  float sc0 = -3.4e38f, sc1 = -3.4e38f;
  for (int p = 0; p < PLc; ++p) {
    size_t o = ((size_t)(b * PLc + p)) * (2 * Cc) + h * HDc + lane;
    float s = wred_add(qs * pkv[o]);
    float sc = s + cpbt[(size_t)rpi[(size_t)n * PLc + p] * 8 + h];
    if ((p & 31) == lane) { if (p < 32) sc0 = sc; else sc1 = sc; }
  }

  // joint softmax over 9 + 64
  float m = -3.4e38f;
#pragma unroll
  for (int l = 0; l < LOCALc; ++l) m = fmaxf(m, aloc[l]);
  m = fmaxf(m, wred_max(fmaxf(sc0, sc1)));
  float den = 0.f;
#pragma unroll
  for (int l = 0; l < LOCALc; ++l) { aloc[l] = expf(aloc[l] - m); den += aloc[l]; }
  float e0 = expf(sc0 - m), e1 = expf(sc1 - m);
  den += wred_add(e0 + e1);
  float inv = 1.0f / den;

  float xl = 0.f;
#pragma unroll
  for (int l = 0; l < LOCALc; ++l) {
    float a = aloc[l] * inv + tl[l] + lbias[h * LOCALc + l];
    xl += a * vloc[l];
  }

  float ap0 = e0 * inv, ap1 = e1 * inv;
  float xp = 0.f;
  for (int p = 0; p < PLc; ++p) {
    float a = __shfl((p < 32) ? ap0 : ap1, p & 31, 32);
    xp += a * pkv[((size_t)(b * PLc + p)) * (2 * Cc) + Cc + h * HDc + lane];
  }

  float outv = (xl + xp) * maskedw[(size_t)(b * Nn + n) * 8 + h];
  outbf[qoff] = (__bf16)outv;
}

// ---------------------------------------------------------------------------
// Host launch
// ---------------------------------------------------------------------------
extern "C" void kernel_launch(void* const* d_in, const int* in_sizes, int n_in,
                              void* d_out, int out_size, void* d_ws, size_t ws_size,
                              hipStream_t stream) {
  (void)in_sizes; (void)n_in; (void)out_size; (void)ws_size;
  const float* x      = (const float*)d_in[0];
  const float* rct    = (const float*)d_in[1];
  const float* q_w    = (const float*)d_in[2];
  const float* q_b    = (const float*)d_in[3];
  const float* kv_w   = (const float*)d_in[4];
  const float* kv_b   = (const float*)d_in[5];
  const float* temp   = (const float*)d_in[6];
  const float* qe     = (const float*)d_in[7];
  const float* rpb    = (const float*)d_in[8];
  const float* ltok   = (const float*)d_in[9];
  const float* lbias  = (const float*)d_in[10];
  const float* cpb1_w = (const float*)d_in[11];
  const float* cpb1_b = (const float*)d_in[12];
  const float* cpb2_w = (const float*)d_in[13];
  const float* cpb2_b = (const float*)d_in[14];
  const float* sr_w   = (const float*)d_in[15];
  const float* sr_b   = (const float*)d_in[16];
  const float* norm_g = (const float*)d_in[17];
  const float* norm_b = (const float*)d_in[18];
  const float* wg_w   = (const float*)d_in[19];
  const float* wg0_w  = (const float*)d_in[20];
  const float* wg1_w  = (const float*)d_in[21];
  const float* proj_w = (const float*)d_in[22];
  const float* proj_b = (const float*)d_in[23];
  const int*   rpi    = (const int*)d_in[24];
  float* out = (float*)d_out;

  char* wsp = (char*)d_ws;
  auto take = [&](size_t bytes) {
    char* p = wsp;
    wsp += (bytes + 255) & ~(size_t)255;
    return p;
  };
  __bf16* xbf    = (__bf16*)take((size_t)BN * Cc * 2);
  __bf16* wqb    = (__bf16*)take((size_t)Cc * Cc * 2);
  __bf16* wkvb   = (__bf16*)take((size_t)2 * Cc * Cc * 2);
  __bf16* wsrb   = (__bf16*)take((size_t)Cc * Cc * 2);
  __bf16* wprb   = (__bf16*)take((size_t)Cc * Cc * 2);
  float*  qbuf   = (float*)take((size_t)BN * Cc * 4);
  float*  qsbuf  = (float*)take((size_t)BN * Cc * 4);
  float*  kvbuf  = (float*)take((size_t)BN * 2 * Cc * 4);
  float*  srbuf  = (float*)take((size_t)BN * Cc * 4);
  float*  xpool  = (float*)take((size_t)Bc * PLc * Cc * 4);
  __bf16* xlnbf  = (__bf16*)take((size_t)Bc * PLc * Cc * 2);
  float*  pkv    = (float*)take((size_t)Bc * PLc * 2 * Cc * 4);
  float*  cpbt   = (float*)take((size_t)TBLc * HEADSc * 4);
  float*  maskw  = (float*)take((size_t)BN * HEADSc * 4);
  __bf16* attnbf = (__bf16*)take((size_t)BN * Cc * 2);

  auto cvt = [&](const float* src, __bf16* dst, int n) {
    cvt_bf16_kernel<<<(n + 255) / 256, 256, 0, stream>>>(src, dst, n);
  };
  cvt(x, xbf, BN * Cc);
  cvt(q_w, wqb, Cc * Cc);
  cvt(kv_w, wkvb, 2 * Cc * Cc);
  cvt(sr_w, wsrb, Cc * Cc);
  cvt(proj_w, wprb, Cc * Cc);

  gates_kernel<<<BN * 32 / 256, 256, 0, stream>>>(x, wg_w, wg0_w, wg1_w, maskw);

  // q = x @ q_w^T + q_b
  wmma_gemm_bf16<Cc><<<dim3(BN / 128, Cc / 64), 256, 0, stream>>>(xbf, wqb, q_b, qbuf, BN, Cc);
  q_post_kernel<<<BN * HEADSc * 32 / 256, 256, 0, stream>>>(qbuf, qsbuf, qe, temp);

  // kv = x @ kv_w^T + kv_b ; L2-normalize k part per head
  wmma_gemm_bf16<Cc><<<dim3(BN / 128, 2 * Cc / 64), 256, 0, stream>>>(xbf, wkvb, kv_b, kvbuf, BN, 2 * Cc);
  kvnorm_kernel<<<BN * HEADSc * 32 / 256, 256, 0, stream>>>(kvbuf, BN);

  // sr branch: GEMM -> gelu + 8x8 pool -> layernorm -> pooled kv GEMM -> norm
  wmma_gemm_bf16<Cc><<<dim3(BN / 128, Cc / 64), 256, 0, stream>>>(xbf, wsrb, sr_b, srbuf, BN, Cc);
  srpool_kernel<<<(Bc * PLc * Cc) / 256, 256, 0, stream>>>(srbuf, xpool);
  ln_kernel<<<Bc * PLc, 256, 0, stream>>>(xpool, norm_g, norm_b, xlnbf);
  wmma_gemm_bf16<Cc><<<dim3(Bc * PLc / 128, 2 * Cc / 64), 256, 0, stream>>>(xlnbf, wkvb, kv_b, pkv,
                                                                            Bc * PLc, 2 * Cc);
  kvnorm_kernel<<<Bc * PLc * HEADSc * 32 / 256, 256, 0, stream>>>(pkv, Bc * PLc);

  cpb_kernel<<<TBLc * HEADSc / 256, 256, 0, stream>>>(rct, cpb1_w, cpb1_b, cpb2_w, cpb2_b, cpbt);

  attn_kernel<<<Bc * HEADSc * Nn * 32 / 256, 256, 0, stream>>>(qbuf, qsbuf, kvbuf, pkv, cpbt,
                                                               rpi, rpb, ltok, lbias, maskw, attnbf);

  // final projection straight into d_out
  wmma_gemm_bf16<Cc><<<dim3(BN / 128, Cc / 64), 256, 0, stream>>>(attnbf, wprb, proj_b, out, BN, Cc);
}
